// NonLocalBlock_19499151524502
// MI455X (gfx1250) — matrix operand order repaired and checked by reading, not compiled, not run on hardware
//
#include <hip/hip_runtime.h>
#include <math.h>
#include <stdint.h>

// ---------------------------------------------------------------------------
// NonLocalBlock for gfx1250 (MI455X): flash-attention fused with WMMA f16,
// KV tiles double-buffered in LDS via async load-to-LDS (ASYNCcnt).
// B=8, C=256, N=4096, Ci=128.
// ---------------------------------------------------------------------------

typedef __attribute__((ext_vector_type(16))) _Float16 v16h;
typedef __attribute__((ext_vector_type(8)))  float    v8f;
typedef __attribute__((ext_vector_type(4)))  int      v4i;
typedef _Float16 half_t;

union Frag16 { v16h h; uint4 q[2]; };

__device__ __forceinline__ v8f wmma32f16(v16h a, v16h b, v8f c) {
  return __builtin_amdgcn_wmma_f32_16x16x32_f16(false, a, false, b, (short)0, c,
                                                false, false);
}

// A-operand (16x32 f16) fragment from K-major storage.
// lane l<16: row M=l, K=0..15; lane l+16: K=16..31 (contiguous 32B).
__device__ __forceinline__ v16h load_a_frag(const half_t* rowptr, int hi) {
  Frag16 f;
  const half_t* p = rowptr + hi * 16;
  f.q[0] = *(const uint4*)(p);
  f.q[1] = *(const uint4*)(p + 8);
  return f.h;
}

// B-operand (32x16 f16) fragment from K-major storage (row = output column).
// lane l<16 holds K=0..15, lane l+16 holds K=16..31 (contiguous 32B).
__device__ __forceinline__ v16h load_b_frag(const half_t* rowptr, int hi) {
  Frag16 f;
  const half_t* p = rowptr + hi * 16;
  f.q[0] = *(const uint4*)(p);
  f.q[1] = *(const uint4*)(p + 8);
  return f.h;
}

__device__ __forceinline__ uint4 pack8_f16(v8f v) {
  union { half_t h[8]; uint4 u; } r;
#pragma unroll
  for (int i = 0; i < 8; ++i) r.h[i] = (half_t)v[i];
  return r.u;
}

__device__ __forceinline__ uint4 shfl_xor16_u4(uint4 v) {
  uint4 r;
  r.x = (unsigned)__shfl_xor((int)v.x, 16, 32);
  r.y = (unsigned)__shfl_xor((int)v.y, 16, 32);
  r.z = (unsigned)__shfl_xor((int)v.z, 16, 32);
  r.w = (unsigned)__shfl_xor((int)v.w, 16, 32);
  return r;
}

// ----- CDNA5 async global->LDS copy (GLOBAL_LOAD_ASYNC_TO_LDS_B128) --------
#if __has_builtin(__builtin_amdgcn_global_load_async_to_lds_b128)
#define HAVE_ASYNC_LDS 1
typedef __attribute__((address_space(1))) v4i* gv4i_ptr;  // global int4*
typedef __attribute__((address_space(3))) v4i* lv4i_ptr;  // LDS int4*
__device__ __forceinline__ void async_cp16(const uint4* g, uint4* l) {
  __builtin_amdgcn_global_load_async_to_lds_b128(
      (gv4i_ptr)(uintptr_t)g,
      (lv4i_ptr)(uint32_t)(uintptr_t)l, 0, 0);
}
#else
__device__ __forceinline__ void async_cp16(const uint4* g, uint4* l) { *l = *g; }
#endif

__device__ __forceinline__ void async_wait_all() {
#if defined(HAVE_ASYNC_LDS)
#if __has_builtin(__builtin_amdgcn_s_wait_asynccnt)
  __builtin_amdgcn_s_wait_asynccnt(0);
#else
  asm volatile("s_wait_asynccnt 0x0" ::: "memory");
#endif
#endif
}

// ---------------------------------------------------------------------------
// Kernel 0: fp32 -> fp16 weight conversion
// ---------------------------------------------------------------------------
__global__ __launch_bounds__(256) void cvt_f16_kernel(const float* __restrict__ src,
                                                      half_t* __restrict__ dst, int n) {
  int i = blockIdx.x * 256 + threadIdx.x;
  if (i < n) dst[i] = (half_t)src[i];
}

// ---------------------------------------------------------------------------
// Kernel 1: theta/phi/g projections.  Out = W[128x256] * X[256xN] + b.
//   theta, phi stored [B][N][128]; g stored [B][128][N].
// ---------------------------------------------------------------------------
#define CPAD 264  // 256 + 8 halfs pad -> conflict-free strided reads

__global__ __launch_bounds__(256) void proj_kernel(
    const float* __restrict__ x,
    const half_t* __restrict__ wth, const half_t* __restrict__ wph,
    const half_t* __restrict__ wg,
    const float* __restrict__ bth, const float* __restrict__ bph,
    const float* __restrict__ bg,
    half_t* __restrict__ th, half_t* __restrict__ ph, half_t* __restrict__ g) {
  __shared__ __align__(16) half_t xt[64 * CPAD];  // x tile transposed: [n][c]
  const int b  = blockIdx.x;
  const int n0 = blockIdx.y * 64;
  const int t  = threadIdx.x;

  // Load + transpose x tile: thread t = channel c, reads 64 n-values.
  const float* xrow = x + ((size_t)b * 256 + t) * 4096 + n0;
#pragma unroll 4
  for (int i = 0; i < 16; ++i) {
    float4 v = ((const float4*)xrow)[i];
    int n = i * 4;
    xt[(n + 0) * CPAD + t] = (half_t)v.x;
    xt[(n + 1) * CPAD + t] = (half_t)v.y;
    xt[(n + 2) * CPAD + t] = (half_t)v.z;
    xt[(n + 3) * CPAD + t] = (half_t)v.w;
  }
  __syncthreads();

  const int wave = t >> 5, lane = t & 31;
  const int l15 = lane & 15, hi = lane >> 4;
  const int o0 = wave * 16;  // 8 waves x 16 = 128 out channels

#pragma unroll 1             // keep VGPR count < 256 (one proj's A-frags live)
  for (int nt = 0; nt < 4; ++nt) {
    v16h bf[8];  // full K=256 of the x-tile columns, reused for all 3 projs
    const half_t* ldsrow = &xt[(nt * 16 + l15) * CPAD];
#pragma unroll
    for (int k = 0; k < 8; ++k) bf[k] = load_b_frag(ldsrow + k * 32, hi);

    const int n = n0 + nt * 16 + l15;
#pragma unroll 1
    for (int pj = 0; pj < 3; ++pj) {
      const half_t* w    = (pj == 0) ? wth : ((pj == 1) ? wph : wg);
      const float*  bias = (pj == 0) ? bth : ((pj == 1) ? bph : bg);
      v8f acc = {};
      const half_t* wrow = w + (o0 + l15) * 256;
#pragma unroll
      for (int k = 0; k < 8; ++k)
        acc = wmma32f16(load_a_frag(wrow + k * 32, hi), bf[k], acc);
      const float4* bp = (const float4*)(bias + o0 + hi * 8);
      float4 b0 = bp[0], b1 = bp[1];
      acc[0] += b0.x; acc[1] += b0.y; acc[2] += b0.z; acc[3] += b0.w;
      acc[4] += b1.x; acc[5] += b1.y; acc[6] += b1.z; acc[7] += b1.w;
      if (pj < 2) {  // theta/phi: [n][o] layout, contiguous 16B per lane
        half_t* out = ((pj == 0) ? th : ph) +
                      ((size_t)b * 4096 + n) * 128 + o0 + hi * 8;
        *(uint4*)out = pack8_f16(acc);
      } else {       // g: [o][n] layout
        half_t* gbp = g + (size_t)b * 128 * 4096;
#pragma unroll
        for (int r = 0; r < 8; ++r)
          gbp[(size_t)(o0 + hi * 8 + r) * 4096 + n] = (half_t)acc[r];
      }
    }
  }
}

// ---------------------------------------------------------------------------
// Kernel 2: flash attention with LDS double buffering + async copies.
//   S^T = Ph^T * Th  (query on the lane -> per-lane online softmax)
//   Y^T += G * P     (exp'd scores repack directly into the B-operand layout)
// WG = 8 waves = 128 queries; KV streamed in 64-key blocks.
// ---------------------------------------------------------------------------
#define PHP 136                 // 128 + 8 pad (halfs)
#define GP  72                  // 64 + 8 pad (halfs)
#define PH_TILE (64 * PHP)      // halfs per Ph buffer
#define G_TILE  (128 * GP)      // halfs per G buffer
#define ATTN_LDS_BYTES (2 * (PH_TILE + G_TILE) * 2)

__global__ __launch_bounds__(256) void attn_kernel(
    const half_t* __restrict__ th, const half_t* __restrict__ ph,
    const half_t* __restrict__ g, half_t* __restrict__ y) {
  extern __shared__ __align__(16) char smem[];
  half_t* sph = (half_t*)smem;                            // 2 x PH_TILE
  half_t* sg  = (half_t*)(smem + 2 * PH_TILE * 2);        // 2 x G_TILE

  const int b = blockIdx.x;
  const int t = threadIdx.x, wave = t >> 5, lane = t & 31;
  const int l15 = lane & 15, hi = lane >> 4;
  const int n = blockIdx.y * 128 + wave * 16 + l15;  // this lane's query

  const half_t* phb = ph + (size_t)b * 4096 * 128;
  const half_t* gb  = g  + (size_t)b * 128 * 4096;

  // cooperative-copy slots for this thread
  const int prow = t >> 2, pseg = t & 3;   // Ph: 64 rows x 4 x 64B
  const int grow = t >> 1, gseg = t & 1;   // G : 128 rows x 2 x 64B

  auto issue_tile = [&](int m0, int buf) {
    const uint4* s1 = (const uint4*)(phb + (size_t)(m0 + prow) * 128 + pseg * 32);
    uint4* d1 = (uint4*)(sph + buf * PH_TILE + prow * PHP + pseg * 32);
    async_cp16(s1 + 0, d1 + 0);
    async_cp16(s1 + 1, d1 + 1);
    async_cp16(s1 + 2, d1 + 2);
    async_cp16(s1 + 3, d1 + 3);
    const uint4* s2 = (const uint4*)(gb + (size_t)grow * 4096 + m0 + gseg * 32);
    uint4* d2 = (uint4*)(sg + buf * G_TILE + grow * GP + gseg * 32);
    async_cp16(s2 + 0, d2 + 0);
    async_cp16(s2 + 1, d2 + 1);
    async_cp16(s2 + 2, d2 + 2);
    async_cp16(s2 + 3, d2 + 3);
  };

  // Resident query (Th) B-operand fragments: K = Ci = 128.
  v16h qf[4];
  const half_t* throw_ = th + ((size_t)b * 4096 + n) * 128;
#pragma unroll
  for (int k = 0; k < 4; ++k) qf[k] = load_b_frag(throw_ + k * 32, hi);

  v8f vz = {};
  v8f yacc[8];
#pragma unroll
  for (int i = 0; i < 8; ++i) yacc[i] = vz;
  float mrun = -INFINITY, lrun = 0.f;

  // prologue: DMA first tile
  issue_tile(0, 0);
  async_wait_all();
  __syncthreads();

  for (int blk = 0; blk < 64; ++blk) {
    const int cur = blk & 1;
    if (blk + 1 < 64) issue_tile((blk + 1) * 64, cur ^ 1);  // overlap DMA

    const half_t* sph_c = sph + cur * PH_TILE;
    const half_t* sg_c  = sg  + cur * G_TILE;

    // S^T sub-tiles: rows m (16 at a time), columns = 16 queries.
    v8f s4[4];
#pragma unroll
    for (int sub = 0; sub < 4; ++sub) {
      v8f acc = vz;
      const half_t* prow_c = &sph_c[(sub * 16 + l15) * PHP];
#pragma unroll
      for (int k = 0; k < 4; ++k)
        acc = wmma32f16(load_a_frag(prow_c + k * 32, hi), qf[k], acc);
      s4[sub] = acc;
    }

    // Online softmax; query n is split across lanes l and l+16.
    float lm = s4[0][0];
#pragma unroll
    for (int sub = 0; sub < 4; ++sub)
#pragma unroll
      for (int r = 0; r < 8; ++r) lm = fmaxf(lm, s4[sub][r]);
    lm = fmaxf(lm, __shfl_xor(lm, 16, 32));
    float mnew = fmaxf(mrun, lm);
    float rescale = __expf(mrun - mnew);
    float ls = 0.f;
#pragma unroll
    for (int sub = 0; sub < 4; ++sub)
#pragma unroll
      for (int r = 0; r < 8; ++r) {
        float e = __expf(s4[sub][r] - mnew);
        s4[sub][r] = e;
        ls += e;
      }
    ls += __shfl_xor(ls, 16, 32);
    lrun = lrun * rescale + ls;
    mrun = mnew;
#pragma unroll
    for (int i = 0; i < 8; ++i)
#pragma unroll
      for (int r = 0; r < 8; ++r) yacc[i][r] *= rescale;

    // Pack P to f16 and swap halves between lane pairs -> B-operand layout.
    uint4 pk0 = pack8_f16(s4[0]), pk1 = pack8_f16(s4[1]);
    uint4 pk2 = pack8_f16(s4[2]), pk3 = pack8_f16(s4[3]);
    uint4 xp0 = shfl_xor16_u4(pk0), xp1 = shfl_xor16_u4(pk1);
    uint4 xp2 = shfl_xor16_u4(pk2), xp3 = shfl_xor16_u4(pk3);
    Frag16 pf0, pf1;                     // K = m0..31 and m32..63
    pf0.q[0] = hi ? xp1 : pk0;           // lo: m0..7   | hi: m16..23
    pf0.q[1] = hi ? pk1 : xp0;           // lo: m8..15  | hi: m24..31
    pf1.q[0] = hi ? xp3 : pk2;           // lo: m32..39 | hi: m48..55
    pf1.q[1] = hi ? pk3 : xp2;           // lo: m40..47 | hi: m56..63

    // Y^T += G * P  (A = G tile rows c, K = m)
#pragma unroll
    for (int ct = 0; ct < 8; ++ct) {
      const half_t* grow_c = &sg_c[(ct * 16 + l15) * GP];
      yacc[ct] = wmma32f16(load_a_frag(grow_c, hi),      pf0.h, yacc[ct]);
      yacc[ct] = wmma32f16(load_a_frag(grow_c + 32, hi), pf1.h, yacc[ct]);
    }

    if (blk + 1 < 64) {
      async_wait_all();   // next tile landed in LDS
      __syncthreads();    // all waves done with cur, next visible to all
    }
  }

  // Finalize: divide by softmax denominator, store y as [B][N][Ci] f16.
  float inv = 1.f / lrun;
  half_t* yrow = y + ((size_t)b * 4096 + n) * 128;
#pragma unroll
  for (int ct = 0; ct < 8; ++ct) {
    v8f v = yacc[ct];
#pragma unroll
    for (int r = 0; r < 8; ++r) v[r] *= inv;
    *(uint4*)(yrow + ct * 16 + hi * 8) = pack8_f16(v);
  }
}

// ---------------------------------------------------------------------------
// Kernel 3: wy = W_w * y  ->  [B][256][N] f32.  (W_b cancels through BN.)
// ---------------------------------------------------------------------------
__global__ __launch_bounds__(256) void outproj_kernel(
    const half_t* __restrict__ y, const half_t* __restrict__ ww,
    float* __restrict__ wy) {
  const int b = blockIdx.x, n0 = blockIdx.y * 64;
  const int t = threadIdx.x, wave = t >> 5, lane = t & 31;
  const int l15 = lane & 15, hi = lane >> 4;
#pragma unroll 1
  for (int nt = 0; nt < 4; ++nt) {
    const int n = n0 + nt * 16 + l15;
    v16h bf[4];
    const half_t* yrow = y + ((size_t)b * 4096 + n) * 128;
#pragma unroll
    for (int k = 0; k < 4; ++k) bf[k] = load_b_frag(yrow + k * 32, hi);
#pragma unroll
    for (int oi = 0; oi < 2; ++oi) {
      const int o0 = (wave * 2 + oi) * 16;
      v8f acc = {};
      const half_t* wrow = ww + (o0 + l15) * 128;
#pragma unroll
      for (int k = 0; k < 4; ++k)
        acc = wmma32f16(load_a_frag(wrow + k * 32, hi), bf[k], acc);
#pragma unroll
      for (int r = 0; r < 8; ++r)
        wy[((size_t)b * 256 + o0 + hi * 8 + r) * 4096 + n] = acc[r];
    }
  }
}

// ---------------------------------------------------------------------------
// Kernel 4: per-channel batch statistics -> fused scale/shift.
// ---------------------------------------------------------------------------
__global__ __launch_bounds__(256) void bnstats_kernel(
    const float* __restrict__ wy, const float* __restrict__ gamma,
    const float* __restrict__ beta, float* __restrict__ scale,
    float* __restrict__ shift) {
  const int c = blockIdx.x, t = threadIdx.x;
  float s = 0.f, s2 = 0.f;
  for (int b = 0; b < 8; ++b) {
    const float* p = wy + ((size_t)b * 256 + c) * 4096;
    for (int i = t; i < 4096; i += 256) {
      float v = p[i];
      s += v;
      s2 += v * v;
    }
  }
  __shared__ float rs[256], rs2[256];
  rs[t] = s; rs2[t] = s2;
  __syncthreads();
  for (int st = 128; st > 0; st >>= 1) {
    if (t < st) { rs[t] += rs[t + st]; rs2[t] += rs2[t + st]; }
    __syncthreads();
  }
  if (t == 0) {
    float mean = rs[0] * (1.f / 32768.f);
    float var  = rs2[0] * (1.f / 32768.f) - mean * mean;
    float r = rsqrtf(var + 1e-5f);
    float sc = gamma[c] * r;
    scale[c] = sc;
    shift[c] = beta[c] - mean * sc;
  }
}

// ---------------------------------------------------------------------------
// Kernel 5: z = wy*scale[c] + shift[c] + x
// ---------------------------------------------------------------------------
__global__ __launch_bounds__(256) void final_kernel(
    const float* __restrict__ wy, const float* __restrict__ x,
    const float* __restrict__ scale, const float* __restrict__ shift,
    float* __restrict__ out) {
  size_t i = ((size_t)blockIdx.x * 256 + threadIdx.x) * 4;
  int c = (int)((i >> 12) & 255);
  float4 w  = *(const float4*)(wy + i);
  float4 xv = *(const float4*)(x + i);
  float sc = scale[c], sh = shift[c];
  float4 o;
  o.x = w.x * sc + sh + xv.x;
  o.y = w.y * sc + sh + xv.y;
  o.z = w.z * sc + sh + xv.z;
  o.w = w.w * sc + sh + xv.w;
  *(float4*)(out + i) = o;
}

// ---------------------------------------------------------------------------
// Host launcher
// ---------------------------------------------------------------------------
extern "C" void kernel_launch(void* const* d_in, const int* in_sizes, int n_in,
                              void* d_out, int out_size, void* d_ws, size_t ws_size,
                              hipStream_t stream) {
  (void)in_sizes; (void)n_in; (void)out_size;
  const float* x     = (const float*)d_in[0];
  const float* g_w   = (const float*)d_in[1];
  const float* g_b   = (const float*)d_in[2];
  const float* th_w  = (const float*)d_in[3];
  const float* th_b  = (const float*)d_in[4];
  const float* ph_w  = (const float*)d_in[5];
  const float* ph_b  = (const float*)d_in[6];
  const float* W_w   = (const float*)d_in[7];
  // d_in[8] = W_b: constant per-channel shift cancels exactly through
  // training-mode BatchNorm -> skipped.
  const float* gamma = (const float*)d_in[9];
  const float* beta  = (const float*)d_in[10];

  // Opt-in for >64KB dynamic LDS (gfx1250 WGP has 320KB). Deterministic,
  // host-side only; safe under graph capture.
  (void)hipFuncSetAttribute((const void*)attn_kernel,
                            hipFuncAttributeMaxDynamicSharedMemorySize,
                            ATTN_LDS_BYTES);

  // workspace layout (bytes)
  const size_t OFF_TH  = 0;                         // [8][4096][128] f16
  const size_t OFF_PH  = 8388608;                   // [8][4096][128] f16
  const size_t OFF_G   = 16777216;                  // [8][128][4096] f16
  const size_t OFF_Y   = 25165824;                  // [8][4096][128] f16
  const size_t OFF_WY  = 33554432;                  // [8][256][4096] f32
  const size_t OFF_WT  = 67108864;                  // theta_w f16
  const size_t OFF_WP  = 67174400;                  // phi_w f16
  const size_t OFF_WG  = 67239936;                  // g_w f16
  const size_t OFF_WW  = 67305472;                  // W_w f16
  const size_t OFF_SC  = 67371008;                  // scale [256] f32
  const size_t OFF_SH  = 67372032;                  // shift [256] f32
  const size_t WS_NEED = 67373056;
  if (ws_size < WS_NEED) return;

  char* ws = (char*)d_ws;
  half_t* thb  = (half_t*)(ws + OFF_TH);
  half_t* phb  = (half_t*)(ws + OFF_PH);
  half_t* gxb  = (half_t*)(ws + OFF_G);
  half_t* yb   = (half_t*)(ws + OFF_Y);
  float*  wyb  = (float*)(ws + OFF_WY);
  half_t* wt16 = (half_t*)(ws + OFF_WT);
  half_t* wp16 = (half_t*)(ws + OFF_WP);
  half_t* wg16 = (half_t*)(ws + OFF_WG);
  half_t* ww16 = (half_t*)(ws + OFF_WW);
  float*  scl  = (float*)(ws + OFF_SC);
  float*  shf  = (float*)(ws + OFF_SH);

  cvt_f16_kernel<<<128, 256, 0, stream>>>(th_w, wt16, 32768);
  cvt_f16_kernel<<<128, 256, 0, stream>>>(ph_w, wp16, 32768);
  cvt_f16_kernel<<<128, 256, 0, stream>>>(g_w,  wg16, 32768);
  cvt_f16_kernel<<<128, 256, 0, stream>>>(W_w,  ww16, 32768);

  proj_kernel<<<dim3(8, 64), 256, 0, stream>>>(x, wt16, wp16, wg16,
                                               th_b, ph_b, g_b, thb, phb, gxb);
  attn_kernel<<<dim3(8, 32), 256, ATTN_LDS_BYTES, stream>>>(thb, phb, gxb, yb);
  outproj_kernel<<<dim3(8, 64), 256, 0, stream>>>(yb, ww16, wyb);
  bnstats_kernel<<<256, 256, 0, stream>>>(wyb, gamma, beta, scl, shf);
  final_kernel<<<8192, 256, 0, stream>>>(wyb, x, scl, shf, (float*)d_out);
}
